// NeuralSymbolicMPWithoutDynamicPruningLayer_58067957842561
// MI455X (gfx1250) — compile-verified, matrix operand order checked by reading, not run on hardware
//
#include <hip/hip_runtime.h>
#include <stdint.h>

// Problem constants (match reference)
#define A_   2
#define B_   128
#define N_   50000
#define D_   512
#define E_   4096
#define NP_  50016          // N padded to multiple of 32 for GEMM2 K-loop
#define CLIPV 1e-14f

typedef __attribute__((ext_vector_type(16))) __bf16 v16bf;
typedef __attribute__((ext_vector_type(8)))  __bf16 v8bf;
typedef __attribute__((ext_vector_type(8)))  float  v8f;
typedef __attribute__((ext_vector_type(4)))  int    v4i;

#if defined(__has_builtin)
#  if __has_builtin(__builtin_amdgcn_global_load_async_to_lds_b128)
#    define HAVE_ASYNC_LDS 1
#  endif
#endif
#ifndef HAVE_ASYNC_LDS
#  define HAVE_ASYNC_LDS 0
#endif

// ---------- helpers ----------
__device__ __forceinline__ unsigned short f2bf(float f) {
    unsigned int u = __float_as_uint(f);
    u += 0x7FFFu + ((u >> 16) & 1u);      // round-to-nearest-even
    return (unsigned short)(u >> 16);
}

// WMMA A/B operand fragment from a row-contiguous bf16 buffer:
// two 16-byte chunks per lane (K 0-7/16-23 for lanes 0-15, 8-15/24-31 for 16-31).
__device__ __forceinline__ v16bf load_frag(const unsigned short* rowp, int k0, int koff) {
    v8bf lo = *reinterpret_cast<const v8bf*>(rowp + k0 + koff);
    v8bf hi = *reinterpret_cast<const v8bf*>(rowp + k0 + 16 + koff);
    return __builtin_shufflevector(lo, hi, 0,1,2,3,4,5,6,7,8,9,10,11,12,13,14,15);
}

__device__ __forceinline__ float block_reduce_sum(float v, float* red) {
    int t = threadIdx.x;
    red[t] = v; __syncthreads();
    for (int s = 128; s > 0; s >>= 1) {
        if (t < s) red[t] += red[t + s];
        __syncthreads();
    }
    float r = red[0]; __syncthreads();
    return r;
}

__device__ __forceinline__ float block_reduce_max(float v, float* red) {
    int t = threadIdx.x;
    red[t] = v; __syncthreads();
    for (int s = 128; s > 0; s >>= 1) {
        if (t < s) red[t] = fmaxf(red[t], red[t + s]);
        __syncthreads();
    }
    float r = red[0]; __syncthreads();
    return r;
}

// ---------- 1. edge scatter: symbolic[ab, dst] += val * head_vector[ab, src] ----------
__global__ void k_scatter(const float* __restrict__ hv, const float* __restrict__ ev,
                          const int* __restrict__ esrc, const int* __restrict__ edst,
                          float* __restrict__ sym) {
    int idx = blockIdx.x * blockDim.x + threadIdx.x;
    if (idx >= A_ * B_ * E_) return;
    int ab  = idx / E_;
    int src = esrc[idx], dst = edst[idx];
    float m = ev[idx] * hv[(size_t)ab * N_ + src];
    atomicAdd(&sym[(size_t)ab * N_ + dst], m);
}

// ---------- 2. tail = head + pred -> bf16 ----------
__global__ void k_tail(const float* __restrict__ he, const float* __restrict__ pe,
                       unsigned short* __restrict__ tailB) {
    int i = blockIdx.x * blockDim.x + threadIdx.x;
    if (i < A_ * B_ * D_) tailB[i] = f2bf(he[i] + pe[i]);
}

// ---------- 3. entity f32[N,D] -> entB bf16[N,D] and entBT bf16[D,NP] (transposed, padded) ----------
__global__ void k_convert(const float* __restrict__ ent,
                          unsigned short* __restrict__ entB,
                          unsigned short* __restrict__ entBT) {
    __shared__ unsigned short tile[32][33];
    int blk = blockIdx.x;
    int tn = blk % 1563, td = blk / 1563;           // 1563 * 32 = 50016 = NP_
    int tx = threadIdx.x & 31, ty = threadIdx.x >> 5;
    int n = tn * 32 + ty, d = td * 32 + tx;
    float v = (n < N_) ? ent[(size_t)n * D_ + d] : 0.0f;
    unsigned short h = f2bf(v);
    if (n < N_) entB[(size_t)n * D_ + d] = h;
    tile[ty][tx] = h;
    __syncthreads();
    int dT = td * 32 + ty, nT = tn * 32 + tx;       // nT < NP_ always
    entBT[(size_t)dT * NP_ + nT] = tile[tx][ty];
}

// ---------- 4. GEMM1: score[256, N] = tailB[256,512] x entB^T ----------
// 512 threads = 16 waves; wave w owns M-tile w; whole block shares 5 N-tiles of B
// staged through LDS (double-buffered, async load-to-LDS when available).
#define G1_TPAD 40                       // padded LDS row stride (bf16 elems) -> conflict-free
#define G1_BUFE (5 * 16 * G1_TPAD)       // 3200 bf16 per buffer

__device__ __forceinline__ void g1_fill(const unsigned short* __restrict__ entB,
                                        unsigned short* lds, int n0, int kk) {
    int t = threadIdx.x;
    if (t < 320) {                       // 80 rows x 4 chunks of 16B
        int nr = t >> 2, c = t & 3;
        const unsigned short* gp = entB + (size_t)(n0 + nr) * D_ + kk * 32 + c * 8;
        unsigned short* lp = lds + nr * G1_TPAD + c * 8;
#if HAVE_ASYNC_LDS
        typedef __attribute__((address_space(1))) v4i* as1_v4i_p;
        typedef __attribute__((address_space(3))) v4i* as3_v4i_p;
        __builtin_amdgcn_global_load_async_to_lds_b128(
            (as1_v4i_p)gp, (as3_v4i_p)lp, 0, 0);
#else
        *reinterpret_cast<v8bf*>(lp) = *reinterpret_cast<const v8bf*>(gp);
#endif
    }
}

__device__ __forceinline__ void g1_sync() {
#if HAVE_ASYNC_LDS
#  if __has_builtin(__builtin_amdgcn_s_wait_asynccnt)
    __builtin_amdgcn_s_wait_asynccnt(0);
#  else
    asm volatile("s_wait_asynccnt 0x0" ::: "memory");
#  endif
#endif
    __syncthreads();
}

__global__ void __launch_bounds__(512)
k_gemm1(const unsigned short* __restrict__ tailB,
        const unsigned short* __restrict__ entB,
        float* __restrict__ score) {
    __shared__ unsigned short ldsB[2][G1_BUFE];
    int lane = threadIdx.x & 31;
    int mt   = threadIdx.x >> 5;         // 0..15 : M-tile
    int n0   = blockIdx.x * 80;          // 5 N-tiles of 16 per block (625 blocks)
    int lm = lane & 15;
    int hl = lane >> 4;
    int koff = hl ? 8 : 0;
    const unsigned short* arow = tailB + (size_t)(mt * 16 + lm) * D_;
    v8f c[5] = {};

    g1_fill(entB, ldsB[0], n0, 0);
    g1_sync();
    for (int kk = 0; kk < D_ / 32; ++kk) {
        int cur = kk & 1;
        if (kk + 1 < D_ / 32) g1_fill(entB, ldsB[(kk + 1) & 1], n0, kk + 1);
        v16bf a = load_frag(arow, kk * 32, koff);
        const unsigned short* bbase = ldsB[cur];
#pragma unroll
        for (int j = 0; j < 5; ++j) {
            const unsigned short* bp = bbase + (j * 16 + lm) * G1_TPAD;
            v8bf blo = *reinterpret_cast<const v8bf*>(bp + koff);
            v8bf bhi = *reinterpret_cast<const v8bf*>(bp + koff + 16);
            v16bf b = __builtin_shufflevector(blo, bhi,
                          0,1,2,3,4,5,6,7,8,9,10,11,12,13,14,15);
            c[j] = __builtin_amdgcn_wmma_f32_16x16x32_bf16(
                       false, a, false, b, (short)0, c[j], false, false);
        }
        g1_sync();
    }
    int rbase = mt * 16 + hl * 8;
#pragma unroll
    for (int j = 0; j < 5; ++j)
#pragma unroll
        for (int i = 0; i < 8; ++i)
            score[(size_t)(rbase + i) * N_ + (n0 + j * 16 + lm)] = c[j][i];
}

// ---------- 5. per-row: clip_norm(symbolic) + softmax(score) -> enhanced=clip_norm(sum), in place ----------
__global__ void k_rownorm(const float* __restrict__ sym, float* __restrict__ score) {
    __shared__ float red[256];
    int r = blockIdx.x;                               // 0..255 (a*B + b)
    const float* srow = sym   + (size_t)r * N_;
    float*       crow = score + (size_t)r * N_;
    int t = threadIdx.x;
    float m = -3.4e38f, ss = 0.0f;
    for (int i = t; i < N_; i += 256) {
        m = fmaxf(m, crow[i]);
        float sv = srow[i];
        ss += (sv < CLIPV) ? 0.0f : sv;
    }
    float M  = block_reduce_max(m, red);
    float SS = fmaxf(CLIPV, block_reduce_sum(ss, red));
    float es = 0.0f;
    for (int i = t; i < N_; i += 256) es += __expf(crow[i] - M);
    float ES = block_reduce_sum(es, red);
    float ts = 0.0f;
    for (int i = t; i < N_; i += 256) {
        float sv = srow[i]; sv = (sv < CLIPV) ? 0.0f : sv;
        float v  = sv / SS + __expf(crow[i] - M) / ES;
        v = (v < CLIPV) ? 0.0f : v;
        crow[i] = v; ts += v;
    }
    float TS = fmaxf(CLIPV, block_reduce_sum(ts, red));
    float inv = 1.0f / TS;
    for (int i = t; i < N_; i += 256) crow[i] *= inv;
}

// ---------- 6. agg[b,:] = clip_norm(prod over A) -> bf16 (padded region pre-zeroed) ----------
__global__ void k_agg(const float* __restrict__ enh, unsigned short* __restrict__ aggB) {
    __shared__ float red[256];
    int b = blockIdx.x;                                // 0..127
    const float* e0 = enh + (size_t)(0 * B_ + b) * N_;
    const float* e1 = enh + (size_t)(1 * B_ + b) * N_;
    int t = threadIdx.x;
    float s = 0.0f;
    for (int i = t; i < N_; i += 256) {
        float p = e0[i] * e1[i]; p = (p < CLIPV) ? 0.0f : p;
        s += p;
    }
    float S = fmaxf(CLIPV, block_reduce_sum(s, red));
    float inv = 1.0f / S;
    unsigned short* orow = aggB + (size_t)b * NP_;
    for (int i = t; i < N_; i += 256) {
        float p = e0[i] * e1[i]; p = (p < CLIPV) ? 0.0f : p;
        orow[i] = f2bf(p * inv);
    }
}

// ---------- 7. GEMM2: out[128,512] = aggB[128,NP] x entBT^T   (split-K, atomic accumulate) ----------
__global__ void k_gemm2(const unsigned short* __restrict__ aggB,
                        const unsigned short* __restrict__ entBT,
                        float* __restrict__ out) {
    int lane = threadIdx.x & 31;
    int w = blockIdx.x * (blockDim.x >> 5) + (threadIdx.x >> 5);  // 0..4095
    int ks = w >> 8;                 // 16 K-slices
    int tq = w & 255;
    int mt = tq >> 5;                // 8 M-tiles (b)
    int nt = tq & 31;                // 32 N-tiles (d)
    int lm = lane & 15, hl = lane >> 4, koff = hl ? 8 : 0;
    const unsigned short* arow = aggB  + (size_t)(mt * 16 + lm) * NP_;
    const unsigned short* brow = entBT + (size_t)(nt * 16 + lm) * NP_;
    const int KT = NP_ / 32;         // 1563
    int kbeg = ks * 98;
    int kend = min(KT, kbeg + 98);
    v8f c = {};
    for (int kk = kbeg; kk < kend; ++kk) {
        int k0 = kk * 32;
        v16bf a = load_frag(arow, k0, koff);
        v16bf b = load_frag(brow, k0, koff);
        c = __builtin_amdgcn_wmma_f32_16x16x32_bf16(
                false, a, false, b, (short)0, c, false, false);
    }
    int r = mt * 16 + hl * 8;
#pragma unroll
    for (int i = 0; i < 8; ++i)
        atomicAdd(&out[(size_t)(r + i) * D_ + nt * 16 + lm], c[i]);
}

// ---------- workspace layout (bytes, all 256-aligned) ----------
#define SYM_OFF    ((size_t)0)
#define SYM_BYTES  ((size_t)A_ * B_ * N_ * 4)            //  51,200,000
#define SCORE_OFF  (SYM_OFF + SYM_BYTES)
#define ENTB_OFF   (SCORE_OFF + SYM_BYTES)
#define ENTB_BYTES ((size_t)N_ * D_ * 2)                 //  51,200,000
#define ENTBT_OFF  (ENTB_OFF + ENTB_BYTES)
#define ENTBT_BYTES ((size_t)D_ * NP_ * 2)               //  51,216,384
#define AGGB_OFF   (ENTBT_OFF + ENTBT_BYTES)
#define AGGB_BYTES ((size_t)B_ * NP_ * 2)                //  12,804,096
#define TAILB_OFF  (AGGB_OFF + AGGB_BYTES)
#define TAILB_BYTES ((size_t)A_ * B_ * D_ * 2)

extern "C" void kernel_launch(void* const* d_in, const int* in_sizes, int n_in,
                              void* d_out, int out_size, void* d_ws, size_t ws_size,
                              hipStream_t stream) {
    (void)in_sizes; (void)n_in; (void)out_size; (void)ws_size;
    const float* ent = (const float*)d_in[0];
    const float* hv  = (const float*)d_in[1];
    const float* he  = (const float*)d_in[2];
    const float* pe  = (const float*)d_in[3];
    const float* ev  = (const float*)d_in[4];
    const int*   esrc = (const int*)d_in[5];
    const int*   edst = (const int*)d_in[6];
    float* out = (float*)d_out;

    char* ws = (char*)d_ws;
    float*          sym   = (float*)(ws + SYM_OFF);
    float*          score = (float*)(ws + SCORE_OFF);
    unsigned short* entB  = (unsigned short*)(ws + ENTB_OFF);
    unsigned short* entBT = (unsigned short*)(ws + ENTBT_OFF);
    unsigned short* aggB  = (unsigned short*)(ws + AGGB_OFF);
    unsigned short* tailB = (unsigned short*)(ws + TAILB_OFF);

    (void)hipMemsetAsync(sym, 0, SYM_BYTES, stream);
    (void)hipMemsetAsync(aggB, 0, AGGB_BYTES, stream);            // zero pads K tail for GEMM2
    (void)hipMemsetAsync(out, 0, (size_t)B_ * D_ * sizeof(float), stream);

    k_scatter<<<(A_ * B_ * E_ + 255) / 256, 256, 0, stream>>>(hv, ev, esrc, edst, sym);
    k_tail<<<(A_ * B_ * D_ + 255) / 256, 256, 0, stream>>>(he, pe, tailB);
    k_convert<<<1563 * 16, 1024, 0, stream>>>(ent, entB, entBT);
    k_gemm1<<<625, 512, 0, stream>>>(tailB, entB, score);         // 16 waves/WG, LDS-staged B
    k_rownorm<<<256, 256, 0, stream>>>(sym, score);
    k_agg<<<128, 256, 0, stream>>>(score, aggB);
    k_gemm2<<<512, 256, 0, stream>>>(aggB, entBT, out);           // 4096 waves, split-K 16
}
